// BertSelfAttention_34024730919135
// MI455X (gfx1250) — matrix operand order repaired and checked by reading, not compiled, not run on hardware
//
#include <hip/hip_runtime.h>

// MI455X / gfx1250 flash-attention BERT self-attention.
// f16 WMMA (v_wmma_f32_16x16x32_f16) with fp32 accumulate + fp32 softmax.
// Workspace use: 24 MB (f16 Q [B,H,S,64], K [B,H,S,64], V^T [B,H,64,S]).

typedef _Float16 f16;
typedef __attribute__((ext_vector_type(16))) _Float16 v16h;
typedef __attribute__((ext_vector_type(8)))  float    v8f;

union V16 { v16h v; float4 f4[2]; };

static constexpr int Bb   = 4;
static constexpr int Ss   = 2048;
static constexpr int HIDc = 512;
static constexpr int NHc  = 8;
static constexpr int HDc  = 64;

// ---------------------------------------------------------------------------
// Kernel 1: fused QKV projection.  grid = (M/128, NH, 3), block = 256 (8 waves)
// Each block: 128 rows x 64 cols (one head) of one projection.
// Q,K -> [B,H,S,64] f16 ; V -> transposed [B,H,64,S] f16.
// ---------------------------------------------------------------------------
__global__ void __launch_bounds__(256)
qkv_proj_kernel(const float* __restrict__ hs,
                const float* __restrict__ Wq, const float* __restrict__ bq,
                const float* __restrict__ Wk, const float* __restrict__ bk,
                const float* __restrict__ Wv, const float* __restrict__ bv,
                f16* __restrict__ Qw, f16* __restrict__ Kw, f16* __restrict__ VTw)
{
    const int z = blockIdx.z;                       // 0:Q 1:K 2:V
    const float* W    = (z == 0) ? Wq : (z == 1) ? Wk : Wv;
    const float* bias = (z == 0) ? bq : (z == 1) ? bk : bv;
    const int h       = blockIdx.y;
    const int rowbase = blockIdx.x * 128;
    const int tid   = threadIdx.x;
    const int wv    = tid >> 5;
    const int lane  = tid & 31;
    const int lhalf = lane >> 4;
    const int lmod  = lane & 15;

    __shared__ f16 As[128 * 40];   // 128x32 A tile, padded stride 40
    __shared__ f16 Bt[64 * 40];    // W tile transposed: Bt[n][k], 64x32

    v8f acc[4] = {};               // 16 rows x 64 cols per wave

    for (int k0 = 0; k0 < HIDc; k0 += 32) {
        // --- stage A tile: 128x32 f32 -> f16 (1024 float4 / 256 threads) ---
        #pragma unroll
        for (int i = 0; i < 4; ++i) {
            int idx = tid + i * 256;               // 0..1023
            int r   = idx >> 3;
            int c4  = (idx & 7) * 4;
            const float* src = hs + (size_t)(rowbase + r) * HIDc + k0 + c4;
            float4 vv = *(const float4*)src;
            if (k0 + 32 < HIDc) __builtin_prefetch(src + 32, 0, 1);
            f16* dst = &As[r * 40 + c4];
            dst[0] = (f16)vv.x; dst[1] = (f16)vv.y;
            dst[2] = (f16)vv.z; dst[3] = (f16)vv.w;
        }
        // --- stage B tile transposed: W[k][n] -> Bt[n][k] -------------------
        #pragma unroll
        for (int i = 0; i < 2; ++i) {
            int idx = tid + i * 256;               // 0..511
            int kk  = idx >> 4;
            int n4  = (idx & 15) * 4;
            const float* src = W + (size_t)(k0 + kk) * HIDc + h * HDc + n4;
            float4 vv = *(const float4*)src;
            if (k0 + 32 < HIDc) __builtin_prefetch(src + 32 * HIDc, 0, 1);
            Bt[(n4 + 0) * 40 + kk] = (f16)vv.x;
            Bt[(n4 + 1) * 40 + kk] = (f16)vv.y;
            Bt[(n4 + 2) * 40 + kk] = (f16)vv.z;
            Bt[(n4 + 3) * 40 + kk] = (f16)vv.w;
        }
        __syncthreads();

        // A fragment: 16x32, row = lane%16, K-chunks split by lane half
        V16 a;
        const f16* ap = &As[(wv * 16 + lmod) * 40 + lhalf * 8];
        a.f4[0] = *(const float4*)ap;
        a.f4[1] = *(const float4*)(ap + 16);

        #pragma unroll
        for (int t = 0; t < 4; ++t) {
            V16 bfr;  // B fragment 32x16: col = lane%16, K-chunks split by lane half
            const f16* bp = &Bt[(t * 16 + lmod) * 40 + lhalf * 8];
            bfr.f4[0] = *(const float4*)bp;
            bfr.f4[1] = *(const float4*)(bp + 16);
            acc[t] = __builtin_amdgcn_wmma_f32_16x16x32_f16(
                false, a.v, false, bfr.v, (short)0, acc[t], false, false);
        }
        __syncthreads();
    }

    // --- epilogue: + bias, cast f16, store -------------------------------
    #pragma unroll
    for (int t = 0; t < 4; ++t) {
        const int   col = t * 16 + lmod;
        const float bb  = bias[h * HDc + col];
        #pragma unroll
        for (int rr = 0; rr < 8; ++rr) {
            int r = rowbase + wv * 16 + lhalf * 8 + rr;   // global GEMM row
            int b = r >> 11;                              // /2048
            int s = r & 2047;
            float val = acc[t][rr] + bb;
            if (z == 2) {   // V transposed: [B,H,64,S]
                VTw[((size_t)(b * NHc + h) * HDc + col) * Ss + s] = (f16)val;
            } else {        // Q or K: [B,H,S,64]
                f16* outp = (z == 0) ? Qw : Kw;
                outp[((size_t)(b * NHc + h) * Ss + s) * HDc + col] = (f16)val;
            }
        }
    }
}

// ---------------------------------------------------------------------------
// Kernel 2: flash attention with symmetric ALiBi.
// grid = (S/64, B*NH), block = 128 (4 waves).  Wave owns 16 query rows.
// K/V tiles (32 keys) staged cooperatively in LDS (double-buffered) so the
// 4 waves share one copy; online softmax in fp32; P re-shaped via LDS.
// ---------------------------------------------------------------------------
__global__ void __launch_bounds__(128)
attn_kernel(const f16* __restrict__ Qw, const f16* __restrict__ Kw,
            const f16* __restrict__ VTw, float* __restrict__ out)
{
    const int bh    = blockIdx.y;          // 0..31
    const int b     = bh >> 3;
    const int h     = bh & 7;
    const int tid   = threadIdx.x;
    const int wv    = tid >> 5;
    const int lane  = tid & 31;
    const int lhalf = lane >> 4;
    const int lmod  = lane & 15;
    const int qrow16 = blockIdx.x * 64 + wv * 16;   // wave's first query row

    const f16* Qbh = Qw  + (size_t)bh * Ss * HDc;
    const f16* Kbh = Kw  + (size_t)bh * Ss * HDc;
    const f16* Vbh = VTw + (size_t)bh * HDc * Ss;   // [64][S]

    __shared__ f16 Ks[2][32 * 72];      // K tile [key][d], stride 72 (16B-aligned rows)
    __shared__ f16 Vs[2][64 * 40];      // V^T tile [d][k], stride 40
    __shared__ f16 plds[4][16 * 40];    // wave-private P transpose buffers

    // Q A-fragments: 16x64 -> two 16x32 chunks (loaded once)
    V16 qa[2];
    {
        const f16* qp = Qbh + (size_t)(qrow16 + lmod) * HDc;
        #pragma unroll
        for (int c = 0; c < 2; ++c) {
            const f16* p = qp + c * 32 + lhalf * 8;
            qa[c].f4[0] = *(const float4*)p;
            qa[c].f4[1] = *(const float4*)(p + 16);
        }
    }

    const float slope = -exp2f(-(float)(h + 1));   // ALiBi slopes for NH=8
    const float scale = 0.125f;                    // 1/sqrt(64)

    float m_run[8], l_run[8];
    v8f   o[4] = {};
    #pragma unroll
    for (int rr = 0; rr < 8; ++rr) { m_run[rr] = -3.0e38f; l_run[rr] = 0.f; }

    for (int it = 0; it < Ss / 32; ++it) {
        const int kbase = it * 32;
        const int buf   = it & 1;

        // ---- cooperative staging of K (32x64) and V^T (64x32) tiles ------
        #pragma unroll
        for (int i = 0; i < 2; ++i) {
            int idx = tid + i * 128;               // 0..255
            {   // K: 32 keys x 8 float4
                int key  = idx >> 3;
                int part = idx & 7;
                const f16* src = Kbh + (size_t)(kbase + key) * HDc + part * 8;
                *(float4*)&Ks[buf][key * 72 + part * 8] = *(const float4*)src;
                if (kbase + 64 < Ss) __builtin_prefetch(src + 64 * HDc, 0, 1);
            }
            {   // V^T: 64 dims x 4 float4
                int dim  = idx >> 2;
                int part = idx & 3;
                const f16* src = Vbh + (size_t)dim * Ss + kbase + part * 8;
                *(float4*)&Vs[buf][dim * 40 + part * 8] = *(const float4*)src;
                if (kbase + 64 < Ss) __builtin_prefetch(src + 64, 0, 1);
            }
        }
        __syncthreads();

        // ---- S = Q K^T for 32 keys (two 16-key column tiles) -------------
        v8f s0 = {}, s1 = {};
        #pragma unroll
        for (int c = 0; c < 2; ++c) {
            V16 kb0, kb1;   // B frag: col=key, K = head-dim chunk, split by lane half
            const f16* p0 = &Ks[buf][(lmod) * 72 + c * 32 + lhalf * 8];
            const f16* p1 = &Ks[buf][(16 + lmod) * 72 + c * 32 + lhalf * 8];
            kb0.f4[0] = *(const float4*)p0; kb0.f4[1] = *(const float4*)(p0 + 16);
            kb1.f4[0] = *(const float4*)p1; kb1.f4[1] = *(const float4*)(p1 + 16);
            s0 = __builtin_amdgcn_wmma_f32_16x16x32_f16(
                false, qa[c].v, false, kb0.v, (short)0, s0, false, false);
            s1 = __builtin_amdgcn_wmma_f32_16x16x32_f16(
                false, qa[c].v, false, kb1.v, (short)0, s1, false, false);
        }

        // ---- scale + ALiBi + online softmax (fp32) -----------------------
        f16* pb = &plds[wv][0];
        #pragma unroll
        for (int rr = 0; rr < 8; ++rr) {
            const float qpos = (float)(qrow16 + lhalf * 8 + rr);
            const float kpos = (float)(kbase + lmod);
            float sv0 = s0[rr] * scale + slope * fabsf(kpos - qpos);
            float sv1 = s1[rr] * scale + slope * fabsf(kpos + 16.0f - qpos);

            float mx = fmaxf(sv0, sv1);
            #pragma unroll
            for (int d = 1; d < 16; d <<= 1) mx = fmaxf(mx, __shfl_xor(mx, d, 32));
            float mn   = fmaxf(m_run[rr], mx);
            float corr = expf(m_run[rr] - mn);
            float p0   = expf(sv0 - mn);
            float p1   = expf(sv1 - mn);
            float rs   = p0 + p1;
            #pragma unroll
            for (int d = 1; d < 16; d <<= 1) rs += __shfl_xor(rs, d, 32);
            l_run[rr] = l_run[rr] * corr + rs;
            m_run[rr] = mn;
            #pragma unroll
            for (int t = 0; t < 4; ++t) o[t][rr] *= corr;

            const int prow = lhalf * 8 + rr;         // P row in 16x32 tile
            pb[prow * 40 + lmod]      = (f16)p0;
            pb[prow * 40 + 16 + lmod] = (f16)p1;
        }
        __syncthreads();

        // ---- P (C-layout) -> A-layout via LDS, then O += P @ V -----------
        V16 pa;
        const f16* pp = &plds[wv][lmod * 40 + lhalf * 8];
        pa.f4[0] = *(const float4*)pp;
        pa.f4[1] = *(const float4*)(pp + 16);

        #pragma unroll
        for (int t = 0; t < 4; ++t) {
            V16 vb;   // B frag from V^T tile: col = head-dim, K = key chunks split by lane half
            const f16* vp = &Vs[buf][(t * 16 + lmod) * 40 + lhalf * 8];
            vb.f4[0] = *(const float4*)vp; vb.f4[1] = *(const float4*)(vp + 16);
            o[t] = __builtin_amdgcn_wmma_f32_16x16x32_f16(
                false, pa.v, false, vb.v, (short)0, o[t], false, false);
        }
    }

    // ---- normalize and write fp32 output [B,S,512] ----------------------
    #pragma unroll
    for (int t = 0; t < 4; ++t) {
        const int col = t * 16 + lmod;
        #pragma unroll
        for (int rr = 0; rr < 8; ++rr) {
            const int q = qrow16 + lhalf * 8 + rr;
            out[((size_t)(b * Ss + q)) * HIDc + h * HDc + col] = o[t][rr] / l_run[rr];
        }
    }
}

// ---------------------------------------------------------------------------
extern "C" void kernel_launch(void* const* d_in, const int* in_sizes, int n_in,
                              void* d_out, int out_size, void* d_ws, size_t ws_size,
                              hipStream_t stream) {
    (void)in_sizes; (void)n_in; (void)out_size; (void)ws_size;
    const float* hs = (const float*)d_in[0];
    const float* Wq = (const float*)d_in[1];
    const float* bq = (const float*)d_in[2];
    const float* Wk = (const float*)d_in[3];
    const float* bk = (const float*)d_in[4];
    const float* Wv = (const float*)d_in[5];
    const float* bv = (const float*)d_in[6];
    float* out = (float*)d_out;

    const size_t per = (size_t)Bb * NHc * Ss * HDc;   // 4M f16 elems = 8 MB
    f16* Qw  = (f16*)d_ws;
    f16* Kw  = Qw + per;
    f16* VTw = Kw + per;

    qkv_proj_kernel<<<dim3(Bb * Ss / 128, NHc, 3), 256, 0, stream>>>(
        hs, Wq, bq, Wk, bk, Wv, bv, Qw, Kw, VTw);
    attn_kernel<<<dim3(Ss / 64, Bb * NHc), 128, 0, stream>>>(Qw, Kw, VTw, out);
}
